// ASTGraphEncoder_55490977465023
// MI455X (gfx1250) — matrix operand order repaired and checked by reading, not compiled, not run on hardware
//
#include <hip/hip_runtime.h>
#include <hip/hip_bf16.h>
#include <math.h>

// ---------------- problem constants ----------------
#define NN      50000
#define EE      400000
#define FNF     128
#define FEF     64
#define HH      128
#define NHEADS  4
#define NLAY    3
#define NBATCH  16
#define ENL     (EE + NN)          // edges + self loops
#define LN_EPS  1e-5f

typedef __attribute__((ext_vector_type(16))) __bf16 v16bf;
typedef __attribute__((ext_vector_type(8)))  float  v8f;

// ---------------- small helpers ----------------
__device__ __forceinline__ void atomicMaxF(float* addr, float v) {
    // int-punned float max: works for all-sign floats when buffer init to -inf
    if (v >= 0.f) atomicMax((int*)addr, __float_as_int(v));
    else          atomicMin((unsigned int*)addr, __float_as_uint(v));
}

__device__ __forceinline__ float waveReduceSum(float v) {
    #pragma unroll
    for (int off = 16; off > 0; off >>= 1) v += __shfl_down(v, off, 32);
    return __shfl(v, 0, 32);
}

__global__ void fill_kernel(float* __restrict__ p, float v, long n) {
    long i = (long)blockIdx.x * blockDim.x + threadIdx.x;
    long st = (long)gridDim.x * blockDim.x;
    for (; i < n; i += st) p[i] = v;
}

// ---------------- WMMA GEMM: OUT[MxC] = X[Mx128] @ W[128xC] (+proj epilogue) ----
// grid.x = M/16 row tiles, grid.y = C/128 col slabs, 256 threads = 8 waves.
// LDS holds fragments PRE-SWIZZLED into the WMMA per-lane register layout
// (ISA 7.12.2), so each lane fetches its 16 bf16 as one 32B contiguous read
// (2x ds_load_b128) instead of 16 scalar ds_load_u16 + v_mov_b16 packing.
__global__ void __launch_bounds__(256)
wmma_gemm_kernel(const float* __restrict__ X, const float* __restrict__ W,
                 float* __restrict__ OUT, int cols,
                 const float* __restrict__ bias,          // nullable (proj only)
                 const float* __restrict__ type_table,    // nullable
                 const int*   __restrict__ node_types)    // nullable
{
    __shared__ v16bf fragA[4][32];        // [kstep][lane]          4 KB
    __shared__ v16bf fragB[4][8][32];     // [kstep][wave][lane]   32 KB

    const int tid   = threadIdx.x;
    const int mbase = blockIdx.x * 16;
    const int cbase = blockIdx.y * 128;

    // ---- stage A (16x128 f32 -> swizzled bf16 fragments) ----
    // A layout: lane = half*16 + m, element j=2v+p <-> k32 = 16*(v>=4) + 8*half + 2*(v&3) + p
    {
        const int r  = tid >> 4;          // row 0..15
        const int k0 = (tid & 15) * 8;    // even k base, 8 contiguous k per thread
        const float* src = X + (long)(mbase + r) * 128 + k0;
        float4 f0 = *(const float4*)(src);
        float4 f1 = *(const float4*)(src + 4);
        float v[8] = { f0.x, f0.y, f0.z, f0.w, f1.x, f1.y, f1.z, f1.w };
        #pragma unroll
        for (int t = 0; t < 4; ++t) {
            int k   = k0 + 2 * t;                      // even
            int ks  = k >> 5, k32 = k & 31;
            int hlf = (k32 >> 3) & 1;
            int vv  = ((k32 & 7) >> 1) + ((k32 >> 4) << 2);
            union { __bf16 h[2]; unsigned u; } pk;
            pk.h[0] = (__bf16)v[2 * t];
            pk.h[1] = (__bf16)v[2 * t + 1];
            *(unsigned*)((__bf16*)&fragA[ks][hlf * 16 + r] + vv * 2) = pk.u;
        }
    }
    // ---- stage B (128x128 f32 -> swizzled bf16 fragments) ----
    // B layout: lane = 16*((k32>>3)&1) + (c&15), element j = (k32&7) + 8*(k32>=16); wave owns 16 cols
    {
        const int cq  = (tid & 31) * 4;   // col quad: lanes sweep a row (coalesced)
        const int kp0 = tid >> 5;         // wave id -> k-pair phase
        #pragma unroll
        for (int it = 0; it < 8; ++it) {
            int k = (kp0 + it * 8) * 2;                // even row 0..126
            const float* s0 = W + (long)k * cols + cbase + cq;
            float4 r0 = *(const float4*)(s0);
            float4 r1 = *(const float4*)(s0 + cols);
            float a0[4] = { r0.x, r0.y, r0.z, r0.w };
            float a1[4] = { r1.x, r1.y, r1.z, r1.w };
            int ks = k >> 5, k32 = k & 31;
            int lbase = ((k32 >> 3) & 1) * 16;
            int jh = ((k32 & 7) >> 1) + ((k32 >> 4) << 2); // packed dword slot
            #pragma unroll
            for (int t = 0; t < 4; ++t) {
                int c = cq + t;
                union { __bf16 h[2]; unsigned u; } pk;
                pk.h[0] = (__bf16)a0[t];               // row k
                pk.h[1] = (__bf16)a1[t];               // row k+1
                ((unsigned*)&fragB[ks][c >> 4][lbase + (c & 15)])[jh] = pk.u;
            }
        }
    }
    __syncthreads();

    const int wave = tid >> 5, lane = tid & 31;
    v8f acc = {};
    #pragma unroll
    for (int ks = 0; ks < 4; ++ks) {
        v16bf a = fragA[ks][lane];          // 32B contiguous -> ds_load_b128 x2
        v16bf b = fragB[ks][wave][lane];
        acc = __builtin_amdgcn_wmma_f32_16x16x32_bf16(
                  false, a, false, b, (short)0, acc, false, false);
    }

    // C layout: VGPR r -> M = r + 8*(lane>=16), N = lane%16
    const int nidx = lane & 15, hlf = lane >> 4;
    #pragma unroll
    for (int r = 0; r < 8; ++r) {
        int row = mbase + r + hlf * 8;
        int col = cbase + wave * 16 + nidx;
        float v = acc[r];
        if (bias) v += bias[col] + type_table[(long)node_types[row] * HH + col];
        OUT[(long)row * cols + col] = v;
    }
}

// ---------------- derived attention vectors ----------------
// vs[l,h,c] = sum_k W_gat[l,c,h*128+k]*att_src[l,h,k]   (same for vd, ve)
__global__ void derive_v_kernel(const float* __restrict__ Wg,
                                const float* __restrict__ Weg,
                                const float* __restrict__ a_s,
                                const float* __restrict__ a_d,
                                const float* __restrict__ a_e,
                                float* __restrict__ vs, float* __restrict__ vd,
                                float* __restrict__ ve)
{
    int id = blockIdx.x * 256 + threadIdx.x;
    if (id >= NLAY * NHEADS * HH) return;
    int l = id >> 9, h = (id >> 7) & 3, c = id & 127;
    const float* wg  = Wg  + ((long)l * HH + c) * (NHEADS * HH) + h * HH;
    const float* weg = Weg + ((long)l * HH + c) * (NHEADS * HH) + h * HH;
    const float* ps  = a_s + (l * NHEADS + h) * HH;
    const float* pd  = a_d + (l * NHEADS + h) * HH;
    const float* pe  = a_e + (l * NHEADS + h) * HH;
    float s1 = 0.f, s2 = 0.f, s3 = 0.f;
    for (int k = 0; k < HH; ++k) {
        s1 += wg[k] * ps[k]; s2 += wg[k] * pd[k]; s3 += weg[k] * pe[k];
    }
    vs[id] = s1; vd[id] = s2; ve[id] = s3;
}

// U[l,h,r] = sum_c We[r,c]*ve[l,h,c];  cbe[l,h] = dot(b_edgeproj, ve[l,h])
__global__ void derive_u_kernel(const float* __restrict__ We,
                                const float* __restrict__ be,
                                const float* __restrict__ ve,
                                float* __restrict__ U, float* __restrict__ cbe)
{
    int id = blockIdx.x * 256 + threadIdx.x;
    if (id < NLAY * NHEADS * FEF) {
        int lh = id >> 6, r = id & 63;
        float s = 0.f;
        for (int c = 0; c < HH; ++c) s += We[(long)r * HH + c] * ve[lh * HH + c];
        U[id] = s;
    } else if (id < NLAY * NHEADS * FEF + NLAY * NHEADS) {
        int j = id - NLAY * NHEADS * FEF;
        float s = 0.f;
        for (int c = 0; c < HH; ++c) s += be[c] * ve[j * HH + c];
        cbe[j] = s;
    }
}

// column sums of edge_features (for self-loop mean edge attr)
__global__ void mean_ef_kernel(const float* __restrict__ ef, float* __restrict__ mef)
{
    int g = blockIdx.x * 256 + threadIdx.x;       // 16384 threads
    int k = g & 63, r0 = g >> 6;                  // 256 row-strided partials per col
    float s = 0.f;
    for (long r = r0; r < EE; r += 256) s += ef[r * FEF + k];
    atomicAdd(&mef[k], s);
}

// mean_e[c] = b_edgeproj[c] + sum_r (mef[r]/E)*We[r,c]
__global__ void mean_e_kernel(const float* __restrict__ mef,
                              const float* __restrict__ We,
                              const float* __restrict__ be,
                              float* __restrict__ me)
{
    int c = threadIdx.x;
    float s = be[c];
    for (int r = 0; r < FEF; ++r) s += (mef[r] * (1.f / EE)) * We[(long)r * HH + c];
    me[c] = s;
}

// loop_ale[l,h] = dot(mean_e, ve[l,h])
__global__ void loop_ale_kernel(const float* __restrict__ me,
                                const float* __restrict__ ve,
                                float* __restrict__ lae)
{
    int j = threadIdx.x;
    if (j >= NLAY * NHEADS) return;
    float s = 0.f;
    for (int c = 0; c < HH; ++c) s += me[c] * ve[j * HH + c];
    lae[j] = s;
}

// al_e[l,e,h] = edge_features[e,:] @ U[l,h,:] + cbe[l,h]   (all 3 layers at once)
__global__ void __launch_bounds__(256)
ale_kernel(const float* __restrict__ ef, const float* __restrict__ U,
           const float* __restrict__ cbe, float* __restrict__ ale)
{
    __shared__ float sU[12][64];
    __shared__ float sC[12];
    int tid = threadIdx.x;
    for (int i = tid; i < 768; i += 256) sU[i >> 6][i & 63] = U[i];
    if (tid < 12) sC[tid] = cbe[tid];
    __syncthreads();
    long e = (long)blockIdx.x * 256 + tid;
    if (e >= EE) return;
    __builtin_prefetch(ef + (e + 256) * FEF, 0, 1);
    float acc[12];
    #pragma unroll
    for (int j = 0; j < 12; ++j) acc[j] = sC[j];
    for (int r = 0; r < FEF; ++r) {
        float xv = ef[e * FEF + r];
        #pragma unroll
        for (int j = 0; j < 12; ++j) acc[j] += xv * sU[j][r];
    }
    #pragma unroll
    for (int l = 0; l < NLAY; ++l)
        #pragma unroll
        for (int h = 0; h < NHEADS; ++h)
            ale[((long)l * EE + e) * NHEADS + h] = acc[l * NHEADS + h];
}

// al_s/al_d: one wave per node, 4 channels/lane, vs/vd for this layer in LDS
__global__ void __launch_bounds__(256)
alsd_kernel(const float* __restrict__ x, const float* __restrict__ vs,
            const float* __restrict__ vd, float* __restrict__ als,
            float* __restrict__ ald)
{
    __shared__ float svs[NHEADS][HH], svd[NHEADS][HH];
    int tid = threadIdx.x;
    for (int i = tid; i < NHEADS * HH; i += 256) {
        svs[i >> 7][i & 127] = vs[i];
        svd[i >> 7][i & 127] = vd[i];
    }
    __syncthreads();
    int node = blockIdx.x * 8 + (tid >> 5);
    if (node >= NN) return;
    int lane = tid & 31, c0 = lane * 4;
    float4 xv = *(const float4*)(x + (long)node * HH + c0);
    #pragma unroll
    for (int h = 0; h < NHEADS; ++h) {
        float ps = xv.x * svs[h][c0] + xv.y * svs[h][c0 + 1] +
                   xv.z * svs[h][c0 + 2] + xv.w * svs[h][c0 + 3];
        float pd = xv.x * svd[h][c0] + xv.y * svd[h][c0 + 1] +
                   xv.z * svd[h][c0 + 2] + xv.w * svd[h][c0 + 3];
        ps = waveReduceSum(ps);
        pd = waveReduceSum(pd);
        if (lane == 0) {
            als[(long)node * NHEADS + h] = ps;
            ald[(long)node * NHEADS + h] = pd;
        }
    }
}

// pass1: leaky-relu(al_s[src]+al_d[dst]+al_e) -> pbuf; segment atomic max
__global__ void __launch_bounds__(256)
pass1_kernel(const int* __restrict__ ei, const float* __restrict__ als,
             const float* __restrict__ ald, const float* __restrict__ ale_l,
             const float* __restrict__ lae_l, float* __restrict__ pbuf,
             float* __restrict__ maxb)
{
    long e = (long)blockIdx.x * 256 + threadIdx.x;
    if (e >= ENL) return;
    int s, d;
    float base[NHEADS];
    if (e < EE) {
        s = ei[e]; d = ei[EE + e];
        #pragma unroll
        for (int h = 0; h < NHEADS; ++h) base[h] = ale_l[e * NHEADS + h];
    } else {
        s = d = (int)(e - EE);
        #pragma unroll
        for (int h = 0; h < NHEADS; ++h) base[h] = lae_l[h];
    }
    #pragma unroll
    for (int h = 0; h < NHEADS; ++h) {
        float a = als[(long)s * NHEADS + h] + ald[(long)d * NHEADS + h] + base[h];
        a = (a > 0.f) ? a : 0.2f * a;
        pbuf[e * NHEADS + h] = a;
        atomicMaxF(&maxb[(long)d * NHEADS + h], a);
    }
}

// pass2: p = exp(a - max[dst]); segment atomic sum
__global__ void __launch_bounds__(256)
pass2_kernel(const int* __restrict__ ei, float* __restrict__ pbuf,
             const float* __restrict__ maxb, float* __restrict__ sumb)
{
    long e = (long)blockIdx.x * 256 + threadIdx.x;
    if (e >= ENL) return;
    int d = (e < EE) ? ei[EE + e] : (int)(e - EE);
    #pragma unroll
    for (int h = 0; h < NHEADS; ++h) {
        float p = __expf(pbuf[e * NHEADS + h] - maxb[(long)d * NHEADS + h]);
        pbuf[e * NHEADS + h] = p;
        atomicAdd(&sumb[(long)d * NHEADS + h], p);
    }
}

// pass3: acc[dst,c] += (alpha/4) * xh[src,h,c]  -- one wave per (edge,head)
__global__ void __launch_bounds__(256)
msg_kernel(const int* __restrict__ ei, const float* __restrict__ pbuf,
           const float* __restrict__ sumb, const float* __restrict__ xh,
           float* __restrict__ acc)
{
    long task = (long)blockIdx.x * 8 + (threadIdx.x >> 5);
    if (task >= (long)ENL * NHEADS) return;
    int lane = threadIdx.x & 31;
    long e = task >> 2;
    int  h = (int)(task & 3);
    int s, d;
    if (e < EE) { s = ei[e]; d = ei[EE + e]; } else { s = d = (int)(e - EE); }
    float w = pbuf[e * NHEADS + h];
    w = w / (sumb[(long)d * NHEADS + h] + 1e-16f) * 0.25f;   // fold head-mean
    float4 xv = *(const float4*)(xh + ((long)s * NHEADS + h) * HH + lane * 4);
    float* dp = acc + (long)d * HH + lane * 4;
    atomicAdd(dp + 0, w * xv.x);
    atomicAdd(dp + 1, w * xv.y);
    atomicAdd(dp + 2, w * xv.z);
    atomicAdd(dp + 3, w * xv.w);
}

// finalize: x = gelu(layernorm(acc + gat_bias + x))  in place, one wave per node
__global__ void __launch_bounds__(256)
finalize_kernel(float* __restrict__ x, const float* __restrict__ acc,
                const float* __restrict__ gbias, const float* __restrict__ g,
                const float* __restrict__ b)
{
    int node = blockIdx.x * 8 + (threadIdx.x >> 5);
    if (node >= NN) return;
    int lane = threadIdx.x & 31, c0 = lane * 4;
    float4 av = *(const float4*)(acc + (long)node * HH + c0);
    float4 rv = *(const float4*)(x + (long)node * HH + c0);
    float t[4];
    t[0] = av.x + gbias[c0 + 0] + rv.x;
    t[1] = av.y + gbias[c0 + 1] + rv.y;
    t[2] = av.z + gbias[c0 + 2] + rv.z;
    t[3] = av.w + gbias[c0 + 3] + rv.w;
    float s = t[0] + t[1] + t[2] + t[3];
    s = waveReduceSum(s);
    float mu = s * (1.f / HH);
    float vv = 0.f;
    #pragma unroll
    for (int i = 0; i < 4; ++i) { float dd = t[i] - mu; vv += dd * dd; }
    vv = waveReduceSum(vv) * (1.f / HH);
    float inv = rsqrtf(vv + LN_EPS);
    float out[4];
    #pragma unroll
    for (int i = 0; i < 4; ++i) {
        float y = (t[i] - mu) * inv * g[c0 + i] + b[c0 + i];
        out[i] = 0.5f * y * (1.f + erff(y * 0.70710678118654752f));
    }
    *(float4*)(x + (long)node * HH + c0) = make_float4(out[0], out[1], out[2], out[3]);
}

// pooling: wave per node, atomic accumulate into [B][4][128] + counts
__global__ void __launch_bounds__(256)
pool_kernel(const float* __restrict__ x, const int* __restrict__ batch,
            const int* __restrict__ ntype, float* __restrict__ pool,
            float* __restrict__ cnt)
{
    int node = blockIdx.x * 8 + (threadIdx.x >> 5);
    if (node >= NN) return;
    int lane = threadIdx.x & 31, c0 = lane * 4;
    int bi = batch[node], t = ntype[node];
    int sec = (t <= 5) ? 0 : ((t <= 20) ? 1 : 2);
    float4 xv = *(const float4*)(x + (long)node * HH + c0);
    float xa[4] = { xv.x, xv.y, xv.z, xv.w };
    float* pb = pool + (long)bi * 4 * HH;
    #pragma unroll
    for (int i = 0; i < 4; ++i) {
        atomicAdd(pb + sec * HH + c0 + i, xa[i]);
        atomicAdd(pb + 3 * HH + c0 + i, xa[i]);
    }
    if (lane == 0) atomicAdd(&cnt[bi], 1.f);
}

__global__ void final_embed_kernel(const float* __restrict__ pool,
                                   const float* __restrict__ cnt,
                                   float* __restrict__ gout)
{
    int id = blockIdx.x * 256 + threadIdx.x;
    if (id >= NBATCH * 4 * HH) return;
    int bi = id >> 9;
    gout[id] = pool[id] / fmaxf(cnt[bi], 1.f);
}

// ---------------- launch ----------------
extern "C" void kernel_launch(void* const* d_in, const int* in_sizes, int n_in,
                              void* d_out, int out_size, void* d_ws, size_t ws_size,
                              hipStream_t stream)
{
    const float* node_features = (const float*)d_in[0];
    const float* edge_features = (const float*)d_in[1];
    const int*   node_types    = (const int*)d_in[2];
    const int*   edge_index    = (const int*)d_in[3];
    const int*   batch         = (const int*)d_in[4];
    const float* W_proj        = (const float*)d_in[5];
    const float* b_proj        = (const float*)d_in[6];
    const float* type_table    = (const float*)d_in[7];
    const float* W_edgeproj    = (const float*)d_in[8];
    const float* b_edgeproj    = (const float*)d_in[9];
    const float* W_gat         = (const float*)d_in[10];
    const float* W_edge_gat    = (const float*)d_in[11];
    const float* att_src       = (const float*)d_in[12];
    const float* att_dst       = (const float*)d_in[13];
    const float* att_edge      = (const float*)d_in[14];
    const float* gat_bias      = (const float*)d_in[15];
    const float* ln_g          = (const float*)d_in[16];
    const float* ln_b          = (const float*)d_in[17];

    float* xcur = (float*)d_out;                       // x lives in-place [N,128]
    float* gemb = (float*)d_out + (long)NN * HH;       // [B, 512]

    float* ws = (float*)d_ws;
    size_t off = 0;
    auto alloc = [&](size_t nf) { float* p = ws + off; off += (nf + 63) & ~(size_t)63; return p; };
    float* XH     = alloc((size_t)NN * NHEADS * HH);   // 25.6M f
    float* ACC    = alloc((size_t)NN * HH);            //  6.4M f
    float* ALS    = alloc((size_t)NN * NHEADS);
    float* ALD    = alloc((size_t)NN * NHEADS);
    float* ALE    = alloc((size_t)NLAY * EE * NHEADS); //  4.8M f
    float* PBUF   = alloc((size_t)ENL * NHEADS);
    float* MAXB   = alloc((size_t)NN * NHEADS);
    float* SUMB   = alloc((size_t)NN * NHEADS);
    float* VS     = alloc(NLAY * NHEADS * HH);
    float* VD     = alloc(NLAY * NHEADS * HH);
    float* VE     = alloc(NLAY * NHEADS * HH);
    float* UU     = alloc(NLAY * NHEADS * FEF);
    float* CBE    = alloc(NLAY * NHEADS);
    float* MEANEF = alloc(FEF);
    float* MEANE  = alloc(HH);
    float* LAE    = alloc(NLAY * NHEADS);
    float* POOL   = alloc((size_t)NBATCH * 4 * HH);
    float* CNT    = alloc(NBATCH);

    // ---- one-time precompute ----
    fill_kernel<<<16, 256, 0, stream>>>(MEANEF, 0.f, FEF);
    fill_kernel<<<32, 256, 0, stream>>>(POOL, 0.f, (long)NBATCH * 4 * HH);
    fill_kernel<<<1, 256, 0, stream>>>(CNT, 0.f, NBATCH);

    derive_v_kernel<<<6, 256, 0, stream>>>(W_gat, W_edge_gat, att_src, att_dst,
                                           att_edge, VS, VD, VE);
    derive_u_kernel<<<4, 256, 0, stream>>>(W_edgeproj, b_edgeproj, VE, UU, CBE);
    mean_ef_kernel<<<64, 256, 0, stream>>>(edge_features, MEANEF);
    mean_e_kernel<<<1, HH, 0, stream>>>(MEANEF, W_edgeproj, b_edgeproj, MEANE);
    loop_ale_kernel<<<1, 32, 0, stream>>>(MEANE, VE, LAE);
    ale_kernel<<<(EE + 255) / 256, 256, 0, stream>>>(edge_features, UU, CBE, ALE);

    // x0 = node_features @ W_proj + b_proj + type_table[node_types]   (WMMA)
    wmma_gemm_kernel<<<dim3(NN / 16, 1), 256, 0, stream>>>(
        node_features, W_proj, xcur, HH, b_proj, type_table, node_types);

    // ---- 3 GAT layers ----
    for (int l = 0; l < NLAY; ++l) {
        // xh = x @ W_gat[l]  (50000x128 @ 128x512, WMMA bf16)
        wmma_gemm_kernel<<<dim3(NN / 16, 4), 256, 0, stream>>>(
            xcur, W_gat + (long)l * HH * NHEADS * HH, XH, NHEADS * HH,
            nullptr, nullptr, nullptr);

        alsd_kernel<<<NN / 8, 256, 0, stream>>>(xcur, VS + l * NHEADS * HH,
                                                VD + l * NHEADS * HH, ALS, ALD);

        fill_kernel<<<256, 256, 0, stream>>>(MAXB, -INFINITY, (long)NN * NHEADS);
        fill_kernel<<<256, 256, 0, stream>>>(SUMB, 0.f, (long)NN * NHEADS);
        fill_kernel<<<2048, 256, 0, stream>>>(ACC, 0.f, (long)NN * HH);

        pass1_kernel<<<(ENL + 255) / 256, 256, 0, stream>>>(
            edge_index, ALS, ALD, ALE + (long)l * EE * NHEADS,
            LAE + l * NHEADS, PBUF, MAXB);
        pass2_kernel<<<(ENL + 255) / 256, 256, 0, stream>>>(
            edge_index, PBUF, MAXB, SUMB);
        msg_kernel<<<(int)(((long)ENL * NHEADS + 7) / 8), 256, 0, stream>>>(
            edge_index, PBUF, SUMB, XH, ACC);

        finalize_kernel<<<NN / 8, 256, 0, stream>>>(
            xcur, ACC, gat_bias + l * HH, ln_g + l * HH, ln_b + l * HH);
    }

    // ---- pooling ----
    pool_kernel<<<NN / 8, 256, 0, stream>>>(xcur, batch, node_types, POOL, CNT);
    final_embed_kernel<<<(NBATCH * 4 * HH + 255) / 256, 256, 0, stream>>>(
        POOL, CNT, gemb);
}
